// get_model_39496519254277
// MI455X (gfx1250) — compile-verified, matrix operand order checked by reading, not run on hardware
//
#include <hip/hip_runtime.h>
#include <math.h>

typedef _Float16 v16h __attribute__((ext_vector_type(16)));
typedef float    v8f  __attribute__((ext_vector_type(8)));

#define EPS_BN 1e-5f

// ---------------------------------------------------------------------------
// FPS: one block per batch. Serial npoint steps, block argmax with
// first-index tie-break. Writes gathered new_xyz directly.
// ---------------------------------------------------------------------------
#define FPS_T 256
__global__ __launch_bounds__(FPS_T) void fps_kernel(
    const float* __restrict__ xyz, float* __restrict__ new_xyz, int N, int S) {
  int b = blockIdx.x;
  const float* P = xyz + (size_t)b * N * 3;
  float* O = new_xyz + (size_t)b * S * 3;
  int t = threadIdx.x;

  float d[32];
#pragma unroll
  for (int i = 0; i < 32; i++) d[i] = 1e10f;

  __shared__ float rv[FPS_T];
  __shared__ int   ri[FPS_T];
  __shared__ int   s_far;
  if (t == 0) s_far = 0;
  __syncthreads();

  for (int it = 0; it < S; it++) {
    int far = s_far;
    float cx = P[far * 3 + 0], cy = P[far * 3 + 1], cz = P[far * 3 + 2];
    if (t == 0) { O[it * 3 + 0] = cx; O[it * 3 + 1] = cy; O[it * 3 + 2] = cz; }
    float best = -1.f; int bi = 0x7fffffff;
#pragma unroll
    for (int i = 0; i < 32; i++) {
      int p = t + i * FPS_T;
      if (p < N) {
        float dx = P[p * 3 + 0] - cx, dy = P[p * 3 + 1] - cy, dz = P[p * 3 + 2] - cz;
        float nd = fminf(d[i], dx * dx + dy * dy + dz * dz);
        d[i] = nd;
        if (nd > best || (nd == best && p < bi)) { best = nd; bi = p; }
      }
    }
    rv[t] = best; ri[t] = bi;
    __syncthreads();
    for (int s2 = FPS_T / 2; s2 > 0; s2 >>= 1) {
      if (t < s2) {
        if (rv[t + s2] > rv[t] || (rv[t + s2] == rv[t] && ri[t + s2] < ri[t])) {
          rv[t] = rv[t + s2]; ri[t] = ri[t + s2];
        }
      }
      __syncthreads();
    }
    if (t == 0) s_far = ri[0];
    __syncthreads();
  }
}

// ---------------------------------------------------------------------------
// Ball-query + gather + recenter + concat feats, f16 output, zero padded to
// Kpad. One wave32 per centroid; ballot/prefix gives first-64-in-order.
// ---------------------------------------------------------------------------
__global__ void group_kernel(const float* __restrict__ xyz,
                             const float* __restrict__ nxyz,
                             const float* __restrict__ feats,
                             _Float16* __restrict__ Xg,
                             int N, int S, int Cf, int Kpad, float r2, int total_c) {
  int gw = blockIdx.x * (blockDim.x >> 5) + (threadIdx.x >> 5);
  if (gw >= total_c) return;
  int lane = threadIdx.x & 31;
  int b = gw / S;
  const float* P = xyz + (size_t)b * N * 3;
  const float* F = feats ? feats + (size_t)b * N * Cf : nullptr;
  size_t base_row = (size_t)gw * 64;
  float nx = nxyz[(size_t)gw * 3 + 0], ny = nxyz[(size_t)gw * 3 + 1], nz = nxyz[(size_t)gw * 3 + 2];

  int total = 0, first = -1;
  for (int b0 = 0; b0 < N; b0 += 32) {   // N is a multiple of 32
    int p = b0 + lane;
    float dx = P[p * 3 + 0] - nx, dy = P[p * 3 + 1] - ny, dz = P[p * 3 + 2] - nz;
    bool in = (dx * dx + dy * dy + dz * dz) <= r2;
    unsigned mask = (unsigned)__ballot(in);
    if (first < 0 && mask != 0u) first = b0 + __ffs((int)mask) - 1;
    int pre = __popc(mask & ((1u << lane) - 1u));
    int slot = total + pre;
    if (in && slot < 64) {
      _Float16* row = Xg + (base_row + slot) * Kpad;
      row[0] = (_Float16)dx; row[1] = (_Float16)dy; row[2] = (_Float16)dz;
      for (int c = 0; c < Cf; c++) row[3 + c] = (_Float16)F[(size_t)p * Cf + c];
      for (int c = 3 + Cf; c < Kpad; c++) row[c] = (_Float16)0.f;
    }
    total += __popc(mask);
    if (total >= 64) break;               // uniform across wave
  }
  if (total < 64) {
    if (first >= 0) {
      float dx = P[first * 3 + 0] - nx, dy = P[first * 3 + 1] - ny, dz = P[first * 3 + 2] - nz;
      for (int j = total + lane; j < 64; j += 32) {
        _Float16* row = Xg + (base_row + j) * Kpad;
        row[0] = (_Float16)dx; row[1] = (_Float16)dy; row[2] = (_Float16)dz;
        for (int c = 0; c < Cf; c++) row[3 + c] = (_Float16)F[(size_t)first * Cf + c];
        for (int c = 3 + Cf; c < Kpad; c++) row[c] = (_Float16)0.f;
      }
    } else {
      for (int j = lane; j < 64; j += 32) {
        _Float16* row = Xg + (base_row + j) * Kpad;
        for (int c = 0; c < Kpad; c++) row[c] = (_Float16)0.f;
      }
    }
  }
}

// sa4: grouped = concat(xyz, feats) over all points. rows = b*Npts + k
__global__ void group_all_kernel(const float* __restrict__ xyz,
                                 const float* __restrict__ feats,
                                 _Float16* __restrict__ Xg,
                                 int Npts, int Cf, int Kpad) {
  int m = blockIdx.x;
  _Float16* row = Xg + (size_t)m * Kpad;
  const float* p = xyz + (size_t)m * 3;
  const float* f = feats + (size_t)m * Cf;
  for (int c = threadIdx.x; c < Kpad; c += blockDim.x) {
    float v = (c < 3) ? p[c] : ((c < 3 + Cf) ? f[c - 3] : 0.f);
    row[c] = (_Float16)v;
  }
}

// ---------------------------------------------------------------------------
// Weight repack: W[Co,Ci] f32 -> WB f16 [K/32][Co][32] with k permutation
// [0..7,16..23,8..15,24..31] so each lane's B fragment is one contiguous read.
// ---------------------------------------------------------------------------
__device__ __forceinline__ int kperm(int p) {
  return p < 8 ? p : (p < 16 ? p + 8 : (p < 24 ? p - 8 : p));
}
__global__ void prep_w_kernel(const float* __restrict__ W, _Float16* __restrict__ WB,
                              int Ci, int Co, int total) {
  int i = blockIdx.x * blockDim.x + threadIdx.x;
  if (i >= total) return;
  int p  = i & 31;
  int o  = (i >> 5) % Co;
  int kt = i / (32 * Co);
  int k  = kt * 32 + kperm(p);
  WB[i] = (k < Ci) ? (_Float16)W[(size_t)o * Ci + k] : (_Float16)0.f;
}

// ---------------------------------------------------------------------------
// WMMA GEMM: Y[M,Co] = A[M,K] * WB + bias.  One 16x16 tile per wave,
// 4 waves / block.  M % 64 == 0, K % 32 == 0, Co % 16 == 0.
// ---------------------------------------------------------------------------
__global__ __launch_bounds__(128) void gemm_kernel(
    const _Float16* __restrict__ A, const _Float16* __restrict__ WB,
    const float* __restrict__ bias, _Float16* __restrict__ Y, int K, int Co) {
  int wave = threadIdx.x >> 5, lane = threadIdx.x & 31;
  int tile_m = (blockIdx.x * 4 + wave) << 4;
  int tile_n = blockIdx.y << 4;
  int hi  = lane >> 4;        // 0 or 1
  int l16 = lane & 15;

  v8f acc = {0.f, 0.f, 0.f, 0.f, 0.f, 0.f, 0.f, 0.f};
  const _Float16* arow = A + (size_t)(tile_m + l16) * K + hi * 8;
  const _Float16* bcol = WB + (size_t)(tile_n + l16) * 32 + hi * 16;
  int Kt = K >> 5;
  for (int kt = 0; kt < Kt; kt++) {
    v16h af, bf;
    const _Float16* ap = arow + kt * 32;
#pragma unroll
    for (int i = 0; i < 8; i++) { af[i] = ap[i]; af[i + 8] = ap[i + 16]; }
    const _Float16* bp = bcol + (size_t)kt * Co * 32;
#pragma unroll
    for (int i = 0; i < 16; i++) bf[i] = bp[i];
    acc = __builtin_amdgcn_wmma_f32_16x16x32_f16(false, af, false, bf,
                                                 (short)0, acc, false, false);
  }
  int col = tile_n + l16;
  float bv = bias ? bias[col] : 0.f;
  int m0 = tile_m + hi * 8;
#pragma unroll
  for (int r = 0; r < 8; r++)
    Y[(size_t)(m0 + r) * Co + col] = (_Float16)(acc[r] + bv);
}

// ---------------------------------------------------------------------------
// BatchNorm (training-mode) support kernels.
// ---------------------------------------------------------------------------
__global__ void zero_kernel(float* p, int n) {
  int i = blockIdx.x * blockDim.x + threadIdx.x;
  if (i < n) p[i] = 0.f;
}

__global__ __launch_bounds__(256) void stats_kernel(
    const _Float16* __restrict__ Y, float* __restrict__ sums,
    float* __restrict__ sumsq, size_t total, int Co) {
  __shared__ float s1[1024], s2[1024];
  for (int c = threadIdx.x; c < Co; c += blockDim.x) { s1[c] = 0.f; s2[c] = 0.f; }
  __syncthreads();
  size_t stride = (size_t)gridDim.x * blockDim.x;
  for (size_t i = (size_t)blockIdx.x * blockDim.x + threadIdx.x; i < total; i += stride) {
    float v = (float)Y[i];
    int c = (int)(i & (size_t)(Co - 1));   // Co is a power of two
    atomicAdd(&s1[c], v);
    atomicAdd(&s2[c], v * v);
  }
  __syncthreads();
  for (int c = threadIdx.x; c < Co; c += blockDim.x) {
    atomicAdd(&sums[c], s1[c]);
    atomicAdd(&sumsq[c], s2[c]);
  }
}

__global__ void finalize_kernel(const float* sums, const float* sumsq,
                                const float* g, const float* be,
                                float* scale, float* shift, float invM, int Co) {
  int c = blockIdx.x * blockDim.x + threadIdx.x;
  if (c < Co) {
    float m = sums[c] * invM;
    float v = fmaxf(sumsq[c] * invM - m * m, 0.f);
    float sc = g[c] * rsqrtf(v + EPS_BN);
    scale[c] = sc;
    shift[c] = be[c] - m * sc;
  }
}

// in-place: Y = relu(scale*Y + shift), f16
__global__ void norm_kernel(_Float16* Y, const float* __restrict__ scale,
                            const float* __restrict__ shift, size_t total, int Co) {
  size_t stride = (size_t)gridDim.x * blockDim.x;
  for (size_t i = (size_t)blockIdx.x * blockDim.x + threadIdx.x; i < total; i += stride) {
    int c = (int)(i & (size_t)(Co - 1));
    float v = fmaxf(scale[c] * (float)Y[i] + shift[c], 0.f);
    Y[i] = (_Float16)v;
  }
}

// pooled[bs,c] = max_k relu(scale*Y[bs,k,c] + shift)
__global__ void pool_kernel(const _Float16* __restrict__ Y, const float* __restrict__ scale,
                            const float* __restrict__ shift, float* __restrict__ out,
                            int Kp, int Co) {
  int bs = blockIdx.x;
  const _Float16* base = Y + (size_t)bs * Kp * Co;
  for (int c = threadIdx.x; c < Co; c += blockDim.x) {
    float sc = scale[c], sh = shift[c];
    float mx = -1e30f;
    for (int k = 0; k < Kp; k++) {
      float v = fmaxf(sc * (float)base[(size_t)k * Co + c] + sh, 0.f);
      mx = fmaxf(mx, v);
    }
    out[(size_t)bs * Co + c] = mx;
  }
}

// FC head: plain f32 linear (leaky_relu slope 1.0 == identity), optional tanh
__global__ void linear_kernel(const float* __restrict__ in, const float* __restrict__ W,
                              const float* __restrict__ bias, float* __restrict__ out,
                              int M, int K, int Co, int do_tanh) {
  int i = blockIdx.x * blockDim.x + threadIdx.x;
  if (i >= M * Co) return;
  int m = i / Co, o = i - m * Co;
  const float* a = in + (size_t)m * K;
  const float* w = W + (size_t)o * K;
  float acc = bias[o];
  for (int k = 0; k < K; k++) acc += a[k] * w[k];
  out[i] = do_tanh ? tanhf(acc) : acc;
}

// ---------------------------------------------------------------------------
// Host orchestration
// ---------------------------------------------------------------------------
extern "C" void kernel_launch(void* const* d_in, const int* in_sizes, int n_in,
                              void* d_out, int out_size, void* d_ws, size_t ws_size,
                              hipStream_t stream) {
  (void)out_size;
  // ---- input pointer mapping (insertion order vs sorted pytree) ----
  const float* pc = nullptr;
  const float *LW[12], *Lb[12], *Lg[12], *Lbe[12], *FW[3], *Fb[3];
  int pstart = 1, fc_first = 0;
  if (n_in >= 55) {
    if (in_sizes[0] == 393216) {
      pc = (const float*)d_in[0];
      pstart = 1;
      fc_first = (in_sizes[1] == 524288) ? 1 : 0;   // 'fc' W1 before sa1 W1(=96)?
    } else {
      pc = (const float*)d_in[54];                  // params before pointcloud
      pstart = 0;
      fc_first = 1;
    }
  } else {
    pc = (const float*)d_in[0];
  }
  int q = pstart;
  if (fc_first) {
    for (int i = 0; i < 3; i++) { FW[i] = (const float*)d_in[q++]; Fb[i] = (const float*)d_in[q++]; }
  }
  for (int l = 0; l < 12; l++) {
    LW[l]  = (const float*)d_in[q++];
    Lb[l]  = (const float*)d_in[q++];
    Lg[l]  = (const float*)d_in[q++];
    Lbe[l] = (const float*)d_in[q++];
  }
  if (!fc_first) {
    for (int i = 0; i < 3; i++) { FW[i] = (const float*)d_in[q++]; Fb[i] = (const float*)d_in[q++]; }
  }

  // ---- per-layer dims ----
  static const int Ci_[12] = {3, 32, 32,   67, 64, 64,   131, 128, 128,  259, 256, 512};
  static const int Co_[12] = {32, 32, 64,  64, 64, 128,  128, 128, 256,  256, 512, 1024};
  static const int Kg_[12] = {32, 32, 32,  96, 64, 64,   160, 128, 128,  288, 256, 512};
  static const int Mst[4]  = {524288, 262144, 131072, 2048};

  // ---- workspace carving ----
  char* w = (char*)d_ws;
  size_t off = 0;
  auto carve = [&](size_t bytes) -> char* {
    off = (off + 255) & ~(size_t)255;
    char* p = w + off;
    off += bytes;
    return p;
  };
  _Float16* BUF0 = (_Float16*)carve(50331648);   // 48 MiB (max grouped / L2-out)
  _Float16* BUF1 = (_Float16*)carve(67108864);   // 64 MiB (max L1/L3-out)
  float* nx1 = (float*)carve(16 * 512 * 3 * 4);
  float* nx2 = (float*)carve(16 * 256 * 3 * 4);
  float* nx3 = (float*)carve(16 * 128 * 3 * 4);
  float* ft1 = (float*)carve((size_t)16 * 512 * 64 * 4);
  float* ft2 = (float*)carve((size_t)16 * 256 * 128 * 4);
  float* ft3 = (float*)carve((size_t)16 * 128 * 256 * 4);
  float* f4  = (float*)carve(16 * 1024 * 4);
  float* h1  = (float*)carve(16 * 512 * 4);
  float* h2  = (float*)carve(16 * 256 * 4);
  float* ssum  = (float*)carve(2048 * 4);        // sums[1024] + sumsq[1024]
  float* scl   = (float*)carve(1024 * 4);
  float* shf   = (float*)carve(1024 * 4);
  _Float16* WB[12];
  for (int l = 0; l < 12; l++) {
    size_t entries = (size_t)(Kg_[l] / 32) * Co_[l] * 32;
    WB[l] = (_Float16*)carve(entries * 2);
  }
  if (off > ws_size) return;   // insufficient workspace

  // ---- repack all weights (every call; deterministic) ----
  for (int l = 0; l < 12; l++) {
    int total = (Kg_[l] / 32) * Co_[l] * 32;
    prep_w_kernel<<<(total + 255) / 256, 256, 0, stream>>>(LW[l], WB[l], Ci_[l], Co_[l], total);
  }

  // ---- shared-MLP runner: Xg lives in BUF0; ping-pong BUF1/BUF0 ----
  auto run_mlp = [&](int stage, int BS, int Kpts, float* pooled) {
    int M = Mst[stage];
    _Float16* bufs[2] = {BUF0, BUF1};
    _Float16* cur = BUF0;
    int cb = 1;
    for (int li = 0; li < 3; li++) {
      int l = stage * 3 + li;
      int Co = Co_[l], K = Kg_[l];
      _Float16* out = bufs[cb];
      dim3 grid(M / 64, Co / 16);
      gemm_kernel<<<grid, 128, 0, stream>>>(cur, WB[l], Lb[l], out, K, Co);
      zero_kernel<<<(2048 + 255) / 256, 256, 0, stream>>>(ssum, 2048);
      size_t total = (size_t)M * Co;
      stats_kernel<<<240, 256, 0, stream>>>(out, ssum, ssum + 1024, total, Co);
      finalize_kernel<<<(Co + 127) / 128, 128, 0, stream>>>(ssum, ssum + 1024, Lg[l], Lbe[l],
                                                            scl, shf, 1.0f / (float)M, Co);
      if (li < 2) {
        int blocks = (int)((total + 255) / 256);
        if (blocks > 65535) blocks = 65535;
        norm_kernel<<<blocks, 256, 0, stream>>>(out, scl, shf, total, Co);
      } else {
        pool_kernel<<<BS, 256, 0, stream>>>(out, scl, shf, pooled, Kpts, Co);
      }
      cur = out;
      cb ^= 1;
    }
  };

  // ---- SA1 ----
  fps_kernel<<<16, FPS_T, 0, stream>>>(pc, nx1, 8192, 512);
  group_kernel<<<(16 * 512 + 3) / 4, 128, 0, stream>>>(pc, nx1, nullptr, BUF0,
                                                       8192, 512, 0, 32, 0.01f, 16 * 512);
  run_mlp(0, 16 * 512, 64, ft1);
  // ---- SA2 ----
  fps_kernel<<<16, FPS_T, 0, stream>>>(nx1, nx2, 512, 256);
  group_kernel<<<(16 * 256 + 3) / 4, 128, 0, stream>>>(nx1, nx2, ft1, BUF0,
                                                       512, 256, 64, 96, 0.04f, 16 * 256);
  run_mlp(1, 16 * 256, 64, ft2);
  // ---- SA3 ----
  fps_kernel<<<16, FPS_T, 0, stream>>>(nx2, nx3, 256, 128);
  group_kernel<<<(16 * 128 + 3) / 4, 128, 0, stream>>>(nx2, nx3, ft2, BUF0,
                                                       256, 128, 128, 160, 0.16f, 16 * 128);
  run_mlp(2, 16 * 128, 64, ft3);
  // ---- SA4 (group all) ----
  group_all_kernel<<<2048, 256, 0, stream>>>(nx3, ft3, BUF0, 128, 256, 288);
  run_mlp(3, 16, 128, f4);
  // ---- FC head (leaky_relu slope 1.0 == identity) ----
  linear_kernel<<<(16 * 512 + 255) / 256, 256, 0, stream>>>(f4, FW[0], Fb[0], h1, 16, 1024, 512, 0);
  linear_kernel<<<(16 * 256 + 255) / 256, 256, 0, stream>>>(h1, FW[1], Fb[1], h2, 16, 512, 256, 0);
  linear_kernel<<<1, 64, 0, stream>>>(h2, FW[2], Fb[2], (float*)d_out, 16, 256, 4, 1);
}
